// STMoE_DistilGPT2_28346784154229
// MI455X (gfx1250) — matrix operand order repaired and checked by reading, not compiled, not run on hardware
//
#include <hip/hip_runtime.h>
#include <cstddef>

// ---------------- model constants ----------------
#define DM    768          // model dim
#define SQ    512          // sequence
#define BBS   4            // batch
#define TOK   (BBS*SQ)     // 2048 tokens
#define HN    12           // heads
#define DHD   64           // head dim
#define FFD   3072         // gpt2 mlp hidden
#define EXPN  8            // experts
#define GH    4096         // 2*DHID (geglu in)
#define HHID  2048         // DHID
#define CAPC  256          // capacity
#define VV    50258        // vocab

typedef __attribute__((ext_vector_type(16))) __bf16 v16bf;
typedef __attribute__((ext_vector_type(8)))  __bf16 v8bf;
typedef __attribute__((ext_vector_type(8)))  float  v8f;
typedef unsigned short ushort_t;

static __device__ __forceinline__ unsigned int f2bf_bits(float f) {
  unsigned int u = __float_as_uint(f);
  return (u + 0x7FFFu + ((u >> 16) & 1u)) >> 16;          // RNE
}

// LDS byte offset of a __shared__ object = low 32 bits of its generic pointer
static __device__ __forceinline__ unsigned lds_off(const void* p) {
  return (unsigned)(unsigned long long)p;
}
// CDNA5 async memory->LDS copy, 16B per lane, tracked by ASYNCcnt
static __device__ __forceinline__ void async_b128(unsigned lds, const void* gaddr) {
  asm volatile("global_load_async_to_lds_b128 %0, %1, off"
               :: "v"(lds), "v"(gaddr) : "memory");
}
static __device__ __forceinline__ void wait_async0() {
  asm volatile("s_wait_asynccnt 0" ::: "memory");
}

// combine two contiguous 8x-bf16 LDS chunks into one 16-wide fragment
static __device__ __forceinline__ v16bf frag16(const unsigned int* p0,
                                               const unsigned int* p1) {
  v8bf a = *(const v8bf*)(const void*)p0;
  v8bf b = *(const v8bf*)(const void*)p1;
  v16bf r;
  #pragma unroll
  for (int i = 0; i < 8; ++i) { r[i] = a[i]; r[8 + i] = b[i]; }
  return r;
}

// fragments + 4 WMMAs (32x32 per wave)
static __device__ __forceinline__ void do_mma(
    const unsigned int (&As)[128][20], const unsigned int (&Bs)[64][20],
    int wm, int wn, int lm, int hl,
    v8f& acc00, v8f& acc01, v8f& acc10, v8f& acc11)
{
  const int am0 = (wm << 5) + lm, am1 = am0 + 16;
  const int bn0 = (wn << 5) + lm, bn1 = bn0 + 16;
  v16bf a0 = frag16(&As[am0][hl * 4], &As[am0][8 + hl * 4]);
  v16bf a1 = frag16(&As[am1][hl * 4], &As[am1][8 + hl * 4]);
  v16bf b0 = frag16(&Bs[bn0][hl * 4], &Bs[bn0][8 + hl * 4]);
  v16bf b1 = frag16(&Bs[bn1][hl * 4], &Bs[bn1][8 + hl * 4]);
  acc00 = __builtin_amdgcn_wmma_f32_16x16x32_bf16(false, a0, false, b0, (short)0, acc00, false, false);
  acc01 = __builtin_amdgcn_wmma_f32_16x16x32_bf16(false, a0, false, b1, (short)0, acc01, false, false);
  acc10 = __builtin_amdgcn_wmma_f32_16x16x32_bf16(false, a1, false, b0, (short)0, acc10, false, false);
  acc11 = __builtin_amdgcn_wmma_f32_16x16x32_bf16(false, a1, false, b1, (short)0, acc11, false, false);
}

// =====================================================================
// Batched tiled GEMM via V_WMMA_F32_16X16X32_BF16 on bf16 operands.
//   C[z][M,N] = act( A[z][M,K] (@ B or B^T) + bias )
// BT=0: B is [K,N] row-major.  BT=1: B is [N,K] row-major (A @ B^T).
// Block: 256 thr = 8 waves; 128x64 C tile; each wave owns 32x32
// (4 accumulators, 4 wmma per K-step).
// Hot loop: A tile (and B^T tile) staged with GLOBAL_LOAD_ASYNC_TO_LDS_B128
// (memory->LDS, no VGPR roundtrip, ASYNCcnt), fenced by s_wait_asynccnt.
// [K,N] B is transposed into LDS with 2 VALU ops per packed pair.
// Edge tiles use clamped (always-legal) ushort loads + select.
// actmode: 1 = gpt2 gelu(tanh) in epilogue.  outbf: 1 = store bf16.
// =====================================================================
template<int BT>
__global__ __launch_bounds__(256) void gemm_wmma(
    const ushort_t* __restrict__ A, const ushort_t* __restrict__ B,
    const float* __restrict__ bias, void* __restrict__ Cv,
    int M, int N, int K, int lda, int ldb, int ldc,
    long sA0, long sA1, long sB0, long sB1, long sC0, long sC1, int nb1,
    int actmode, int outbf)
{
  const int z  = blockIdx.z;
  const int z0 = z / nb1, z1 = z % nb1;
  A += (long)z0 * sA0 + (long)z1 * sA1;
  B += (long)z0 * sB0 + (long)z1 * sB1;
  const long cb = (long)z0 * sC0 + (long)z1 * sC1;

  // packed bf16 pairs, row stride padded to 20 dwords (80B: bank-spread,
  // still 16B-aligned for b128 fragment loads)
  __shared__ unsigned int As[128][20];   // [m][k/2]
  __shared__ unsigned int Bs[64][20];    // [n][k/2]

  const int tid   = threadIdx.x;
  const int lane  = tid & 31;
  const int wv    = tid >> 5;
  const int wm    = wv & 3;              // 4 row tiles of 32
  const int wn    = wv >> 2;             // 2 col tiles of 32
  const int lm    = lane & 15;
  const int hl    = lane >> 4;           // lane half
  const int tileM = blockIdx.y << 7;
  const int tileN = blockIdx.x << 6;

  v8f acc00 = {0.f,0.f,0.f,0.f,0.f,0.f,0.f,0.f};
  v8f acc01 = acc00, acc10 = acc00, acc11 = acc00;

  const bool fastTile = ((tileM + 128) <= M) && ((tileN + 64) <= N);
  const int  kfull    = K & ~31;

  int kt = 0;
  if (fastTile) {
    // ---------------- hot loop: async LDS staging, no bounds tests ----------------
    const int rA  = tid >> 1;
    const int kcA = (tid & 1) << 4;
    const unsigned ldsA = lds_off(&As[rA][kcA >> 1]);
    const ushort_t* gA = A + (long)(tileM + rA) * lda + kcA;
    for (; kt < kfull; kt += 32) {
      async_b128(ldsA,      gA + kt);
      async_b128(ldsA + 16, gA + kt + 8);
      if (BT) {
        const int r  = tid >> 2;
        const int kc = (tid & 3) << 3;
        async_b128(lds_off(&Bs[r][kc >> 1]), B + (long)(tileN + r) * ldb + kt + kc);
        wait_async0();
      } else {  // B[K,N]: interleave two adjacent-k bf16 rows into packed pairs
        wait_async0();
        const int kp = (tid & 15) << 1;
        const int n4 = (tid >> 4) << 2;
        const uint2 a = *(const uint2*)(const void*)(B + (long)(kt + kp)     * ldb + tileN + n4);
        const uint2 b = *(const uint2*)(const void*)(B + (long)(kt + kp + 1) * ldb + tileN + n4);
        Bs[n4 + 0][kp >> 1] = (a.x & 0xffffu) | (b.x << 16);
        Bs[n4 + 1][kp >> 1] = (a.x >> 16)     | (b.x & 0xffff0000u);
        Bs[n4 + 2][kp >> 1] = (a.y & 0xffffu) | (b.y << 16);
        Bs[n4 + 3][kp >> 1] = (a.y >> 16)     | (b.y & 0xffff0000u);
      }
      __syncthreads();
      do_mma(As, Bs, wm, wn, lm, hl, acc00, acc01, acc10, acc11);
      __syncthreads();
    }
  }
  // ---------------- generic tail / edge-tile loop (clamped) ----------------
  for (; kt < K; kt += 32) {
    { // stage A
      const int r  = tid >> 1;
      const int kc = (tid & 1) << 4;
      const int gm = tileM + r;
      const int gmc = gm < M ? gm : (M - 1);
      unsigned int pk[8];
      #pragma unroll
      for (int i = 0; i < 8; ++i) {
        const int k0 = kt + kc + 2 * i, k1 = k0 + 1;
        const unsigned v0 = A[(long)gmc * lda + (k0 < K ? k0 : K - 1)];
        const unsigned v1 = A[(long)gmc * lda + (k1 < K ? k1 : K - 1)];
        const unsigned u0 = (gm < M && k0 < K) ? v0 : 0u;
        const unsigned u1 = (gm < M && k1 < K) ? v1 : 0u;
        pk[i] = u0 | (u1 << 16);
      }
      unsigned int* dst = &As[r][kc >> 1];
      *(uint4*)(void*)dst       = make_uint4(pk[0], pk[1], pk[2], pk[3]);
      *(uint4*)(void*)(dst + 4) = make_uint4(pk[4], pk[5], pk[6], pk[7]);
    }
    if (BT) {
      const int r  = tid >> 2;
      const int kc = (tid & 3) << 3;
      const int gn = tileN + r;
      const int gnc = gn < N ? gn : (N - 1);
      unsigned int pk[4];
      #pragma unroll
      for (int i = 0; i < 4; ++i) {
        const int k0 = kt + kc + 2 * i, k1 = k0 + 1;
        const unsigned v0 = B[(long)gnc * ldb + (k0 < K ? k0 : K - 1)];
        const unsigned v1 = B[(long)gnc * ldb + (k1 < K ? k1 : K - 1)];
        const unsigned u0 = (gn < N && k0 < K) ? v0 : 0u;
        const unsigned u1 = (gn < N && k1 < K) ? v1 : 0u;
        pk[i] = u0 | (u1 << 16);
      }
      *(uint4*)(void*)&Bs[r][kc >> 1] = make_uint4(pk[0], pk[1], pk[2], pk[3]);
    } else {
      const int kp = (tid & 15) << 1;
      const int n4 = (tid >> 4) << 2;
      const int gk0 = kt + kp, gk1 = kt + kp + 1;
      const int g0c = gk0 < K ? gk0 : (K - 1);
      const int g1c = gk1 < K ? gk1 : (K - 1);
      #pragma unroll
      for (int i = 0; i < 4; ++i) {
        const int gn  = tileN + n4 + i;
        const int gnc = gn < N ? gn : (N - 1);
        const unsigned v0 = B[(long)g0c * ldb + gnc];
        const unsigned v1 = B[(long)g1c * ldb + gnc];
        const unsigned u0 = (gn < N && gk0 < K) ? v0 : 0u;
        const unsigned u1 = (gn < N && gk1 < K) ? v1 : 0u;
        Bs[n4 + i][kp >> 1] = u0 | (u1 << 16);
      }
    }
    __syncthreads();
    do_mma(As, Bs, wm, wn, lm, hl, acc00, acc01, acc10, acc11);
    __syncthreads();
  }

  // ---- epilogue (C/D layout: col = lm, rows = hl*8 + r) ----
  const int rowb0 = tileM + (wm << 5) + hl * 8;
  const int rowb1 = rowb0 + 16;
  const int col0  = tileN + (wn << 5) + lm;
  const int col1  = col0 + 16;
  float* Cf = (float*)Cv;
  ushort_t* Ch = (ushort_t*)Cv;
  auto fin = [&](float v) -> float {
    if (actmode == 1)
      v = 0.5f * v * (1.f + tanhf(0.7978845608f * (v + 0.044715f * v * v * v)));
    return v;
  };
  auto st = [&](long idx, float v) {
    if (outbf) Ch[idx] = (ushort_t)f2bf_bits(v);
    else       Cf[idx] = v;
  };
  if (col0 < N) {
    const float bv = bias ? bias[col0] : 0.f;
    #pragma unroll
    for (int r = 0; r < 8; ++r) {
      const int ra = rowb0 + r, rb = rowb1 + r;
      if (ra < M) st(cb + (long)ra * ldc + col0, fin(acc00[r] + bv));
      if (rb < M) st(cb + (long)rb * ldc + col0, fin(acc10[r] + bv));
    }
  }
  if (col1 < N) {
    const float bv = bias ? bias[col1] : 0.f;
    #pragma unroll
    for (int r = 0; r < 8; ++r) {
      const int ra = rowb0 + r, rb = rowb1 + r;
      if (ra < M) st(cb + (long)ra * ldc + col1, fin(acc01[r] + bv));
      if (rb < M) st(cb + (long)rb * ldc + col1, fin(acc11[r] + bv));
    }
  }
}

// ---------------- fp32 -> bf16 conversion ----------------
__global__ void k_f2bf(const float* __restrict__ s, ushort_t* __restrict__ d, long n) {
  long i = (long)blockIdx.x * 256 + threadIdx.x;
  if (i < n) d[i] = (ushort_t)f2bf_bits(s[i]);
}

// ---------------- LayerNorm / RMSNorm (one block per row, dual output) ----------------
__global__ __launch_bounds__(256) void norm_kernel(
    const float* __restrict__ x, const float* __restrict__ g,
    const float* __restrict__ b, float* __restrict__ yf,
    ushort_t* __restrict__ yh, int d, int isrms)
{
  const long row = blockIdx.x;
  const float* xr = x + row * d;
  __shared__ float sh1[256], sh2[256];
  float s = 0.f, s2 = 0.f;
  for (int i = threadIdx.x; i < d; i += 256) { float v = xr[i]; s += v; s2 += v * v; }
  sh1[threadIdx.x] = s; sh2[threadIdx.x] = s2;
  __syncthreads();
  for (int o = 128; o > 0; o >>= 1) {
    if (threadIdx.x < o) { sh1[threadIdx.x] += sh1[threadIdx.x + o];
                           sh2[threadIdx.x] += sh2[threadIdx.x + o]; }
    __syncthreads();
  }
  const float mean = sh1[0] / d, ms = sh2[0] / d;
  float r;
  if (isrms) r = rsqrtf(ms + 1e-6f);
  else       r = rsqrtf(ms - mean * mean + 1e-5f);
  for (int i = threadIdx.x; i < d; i += 256) {
    const float v = isrms ? (xr[i] * r * g[i]) : ((xr[i] - mean) * r * g[i] + b[i]);
    if (yf) yf[row * d + i] = v;
    if (yh) yh[row * d + i] = (ushort_t)f2bf_bits(v);
  }
}

// ---------------- causal softmax: sc[BH*S, S] fp32 -> p16 bf16 ----------------
__global__ __launch_bounds__(256) void attn_softmax(float* __restrict__ sc,
                                                    ushort_t* __restrict__ p16)
{
  const long row = blockIdx.x;
  const int q = (int)(row & (SQ - 1));
  float* p = sc + row * SQ;
  ushort_t* po = p16 + row * SQ;
  __shared__ float sh[256];
  float mx = -3.0e38f;
  for (int j = threadIdx.x; j < SQ; j += 256)
    if (j <= q) mx = fmaxf(mx, p[j] * 0.125f);
  sh[threadIdx.x] = mx; __syncthreads();
  for (int o = 128; o > 0; o >>= 1) {
    if (threadIdx.x < o) sh[threadIdx.x] = fmaxf(sh[threadIdx.x], sh[threadIdx.x + o]);
    __syncthreads();
  }
  mx = sh[0]; __syncthreads();
  float s = 0.f;
  for (int j = threadIdx.x; j < SQ; j += 256) {
    const float e = (j <= q) ? expf(p[j] * 0.125f - mx) : 0.f;
    p[j] = e; s += e;
  }
  sh[threadIdx.x] = s; __syncthreads();
  for (int o = 128; o > 0; o >>= 1) {
    if (threadIdx.x < o) sh[threadIdx.x] += sh[threadIdx.x + o];
    __syncthreads();
  }
  const float inv = 1.f / sh[0];
  for (int j = threadIdx.x; j < SQ; j += 256)
    po[j] = (ushort_t)f2bf_bits(p[j] * inv);
}

// ---------------- embedding ----------------
__global__ void embed_kernel(const int* __restrict__ ids, const float* __restrict__ wte,
                             const float* __restrict__ wpe, float* __restrict__ x)
{
  const int tk = blockIdx.x;
  const int s  = tk & (SQ - 1);
  const long id = (long)ids[tk];
  const float* te = wte + id * DM;
  const float* pe = wpe + (long)s * DM;
  float* xr = x + (long)tk * DM;
  for (int d = threadIdx.x; d < DM; d += blockDim.x) xr[d] = te[d] + pe[d];
}

// ---------------- elementwise ----------------
__global__ void k_zero(float* p, long n) {
  long i = (long)blockIdx.x * 256 + threadIdx.x; if (i < n) p[i] = 0.f;
}
__global__ void k_add(float* a, const float* b, long n) {
  long i = (long)blockIdx.x * 256 + threadIdx.x; if (i < n) a[i] += b[i];
}
__global__ void k_add2(float* c, const float* a, const float* b, long n) {
  long i = (long)blockIdx.x * 256 + threadIdx.x; if (i < n) c[i] = a[i] + b[i];
}
// geglu over rows of 2*HHID: out(bf16) = a * gelu_exact(g)
__global__ void k_geglu(const float* __restrict__ t, ushort_t* __restrict__ o, long rows) {
  long i = (long)blockIdx.x * 256 + threadIdx.x;
  if (i < rows * (long)HHID) {
    const long m = i >> 11;               // / HHID
    const int  j = (int)(i & (HHID - 1));
    const float a = t[m * (2L * HHID) + j];
    const float g = t[m * (2L * HHID) + HHID + j];
    o[i] = (ushort_t)f2bf_bits(a * (g * 0.5f * (1.f + erff(g * 0.70710678f))));
  }
}
__global__ void k_set(float* dst, const float* src) {
  if (threadIdx.x == 0 && blockIdx.x == 0) dst[0] = src[0];
}

// ---------------- MoE gating (serial per batch, matches jnp argmax/cumsum) ----------------
__global__ void moe_gate(const float* __restrict__ logits,
                         int* __restrict__ e1, int* __restrict__ p1, float* __restrict__ g1,
                         int* __restrict__ e2, int* __restrict__ p2, float* __restrict__ g2,
                         float* __restrict__ m1cnt, float* __restrict__ psum,
                         float* __restrict__ zsum)
{
  if (threadIdx.x != 0) return;
  const int b = blockIdx.x;
  int cnt1[EXPN], cnt2[EXPN]; float ps[EXPN];
  for (int e = 0; e < EXPN; ++e) { cnt1[e] = 0; cnt2[e] = 0; ps[e] = 0.f; }
  float zacc = 0.f;
  for (int t = 0; t < SQ; ++t) {
    const int tk = b * SQ + t;
    const float* lg = logits + (long)tk * EXPN;
    float mx = lg[0];
    for (int e = 1; e < EXPN; ++e) mx = fmaxf(mx, lg[e]);
    float pr[EXPN]; float sum = 0.f;
    for (int e = 0; e < EXPN; ++e) { pr[e] = expf(lg[e] - mx); sum += pr[e]; }
    const float inv = 1.f / sum;
    for (int e = 0; e < EXPN; ++e) { pr[e] *= inv; ps[e] += pr[e]; }
    const float lse = logf(sum) + mx;
    zacc += lse * lse;
    int i1 = 0; float b1v = pr[0];
    for (int e = 1; e < EXPN; ++e) if (pr[e] > b1v) { b1v = pr[e]; i1 = e; }
    int i2 = (i1 == 0) ? 1 : 0; float b2v = pr[i2];
    for (int e = 0; e < EXPN; ++e)
      if (e != i1 && pr[e] > b2v) { b2v = pr[e]; i2 = e; }
    const float denom = b1v + b2v + 1e-9f;
    const float gg1 = b1v / denom;
    const int ks = (b2v > 0.2f) ? 1 : 0;
    const float gg2 = ks ? (b2v / denom) : 0.f;
    e1[tk] = i1; p1[tk] = cnt1[i1]; g1[tk] = gg1; cnt1[i1]++;
    e2[tk] = i2; g2[tk] = gg2; p2[tk] = CAPC;       // invalid by default
  }
  for (int t = 0; t < SQ; ++t) {                    // second-choice positions
    const int tk = b * SQ + t;
    if (g2[tk] > 0.f) {
      const int ee = e2[tk];
      p2[tk] = cnt1[ee] + cnt2[ee];
      cnt2[ee]++;
    }
  }
  for (int e = 0; e < EXPN; ++e) {
    atomicAdd(&m1cnt[e], (float)cnt1[e]);
    atomicAdd(&psum[e], ps[e]);
  }
  atomicAdd(zsum, zacc);
}

__global__ void moe_scatter(const ushort_t* __restrict__ z,
                            const int* __restrict__ e1, const int* __restrict__ p1,
                            const int* __restrict__ e2, const int* __restrict__ p2,
                            const float* __restrict__ g2, ushort_t* __restrict__ xin)
{
  const int tk = blockIdx.x;
  const int b = tk >> 9;
  const ushort_t* zr = z + (long)tk * DM;
  const int q1 = p1[tk];
  if (q1 < CAPC) {
    ushort_t* dst = xin + (((long)e1[tk] * BBS + b) * CAPC + q1) * DM;
    for (int d = threadIdx.x; d < DM; d += blockDim.x) dst[d] = zr[d];
  }
  if (g2[tk] > 0.f) {
    const int q2 = p2[tk];
    if (q2 < CAPC) {
      ushort_t* dst = xin + (((long)e2[tk] * BBS + b) * CAPC + q2) * DM;
      for (int d = threadIdx.x; d < DM; d += blockDim.x) dst[d] = zr[d];
    }
  }
}

__global__ void moe_gather(float* __restrict__ x1, const float* __restrict__ eout,
                           const int* __restrict__ e1, const int* __restrict__ p1,
                           const float* __restrict__ g1, const int* __restrict__ e2,
                           const int* __restrict__ p2, const float* __restrict__ g2)
{
  const int tk = blockIdx.x;
  const int b = tk >> 9;
  const int q1 = p1[tk] < CAPC ? p1[tk] : CAPC - 1;
  const int q2 = p2[tk] < CAPC ? p2[tk] : CAPC - 1;
  const float w1 = (p1[tk] < CAPC) ? g1[tk] : 0.f;
  const float w2 = (g2[tk] > 0.f && p2[tk] < CAPC) ? g2[tk] : 0.f;
  const float* s1 = eout + (((long)e1[tk] * BBS + b) * CAPC + q1) * DM;
  const float* s2 = eout + (((long)e2[tk] * BBS + b) * CAPC + q2) * DM;
  float* xr = x1 + (long)tk * DM;
  for (int d = threadIdx.x; d < DM; d += blockDim.x)
    xr[d] += w1 * s1[d] + w2 * s2[d];
}

__global__ void moe_aux(const float* m1cnt, const float* psum, const float* zsum, float* aux)
{
  if (threadIdx.x == 0 && blockIdx.x == 0) {
    const float T = (float)TOK;
    float s = 0.f;
    for (int e = 0; e < EXPN; ++e) s += (m1cnt[e] / T) * (psum[e] / T);
    const float balance = s * (float)EXPN;   // mean_e(d*p) * E*E = (s/E)*E*E
    const float zl = zsum[0] / T;
    aux[0] += 0.01f * balance + 0.001f * zl;
  }
}

// ---------------- host-side helpers ----------------
static inline dim3 eg(long n) { return dim3((unsigned)((n + 255) / 256)); }

static void gemm_launch(hipStream_t st, bool bt,
                        const ushort_t* A, const ushort_t* B, const float* bias, void* C,
                        int M, int N, int K, int lda, int ldb, int ldc,
                        int nb0 = 1, int nb1 = 1,
                        long sA0 = 0, long sA1 = 0, long sB0 = 0, long sB1 = 0,
                        long sC0 = 0, long sC1 = 0, int actmode = 0, int outbf = 0)
{
  dim3 grid((N + 63) / 64, (M + 127) / 128, nb0 * nb1);
  if (bt)
    gemm_wmma<1><<<grid, dim3(256), 0, st>>>(A, B, bias, C, M, N, K, lda, ldb, ldc,
                                             sA0, sA1, sB0, sB1, sC0, sC1, nb1, actmode, outbf);
  else
    gemm_wmma<0><<<grid, dim3(256), 0, st>>>(A, B, bias, C, M, N, K, lda, ldb, ldc,
                                             sA0, sA1, sB0, sB1, sC0, sC1, nb1, actmode, outbf);
}

extern "C" void kernel_launch(void* const* d_in, const int* in_sizes, int n_in,
                              void* d_out, int out_size, void* d_ws, size_t ws_size,
                              hipStream_t stream)
{
  (void)in_sizes; (void)n_in; (void)out_size; (void)ws_size;
  const int*   ids    = (const int*)d_in[0];
  const float* wte    = (const float*)d_in[2];
  const float* wpe    = (const float*)d_in[3];
  const float* ln1_g  = (const float*)d_in[4];
  const float* ln1_b  = (const float*)d_in[5];
  const float* w_qkv  = (const float*)d_in[6];
  const float* b_qkv  = (const float*)d_in[7];
  const float* w_ao   = (const float*)d_in[8];
  const float* b_ao   = (const float*)d_in[9];
  const float* ln2_g  = (const float*)d_in[10];
  const float* ln2_b  = (const float*)d_in[11];
  const float* w_fc   = (const float*)d_in[12];
  const float* b_fc   = (const float*)d_in[13];
  const float* w_mp   = (const float*)d_in[14];
  const float* b_mp   = (const float*)d_in[15];
  const float* lnf_g  = (const float*)d_in[16];
  const float* lnf_b  = (const float*)d_in[17];
  const float* ffb_g  = (const float*)d_in[18];
  const float* ffb_w1 = (const float*)d_in[19];
  const float* ffb_b1 = (const float*)d_in[20];
  const float* ffb_w2 = (const float*)d_in[21];
  const float* ffb_b2 = (const float*)d_in[22];
  const float* moe_g  = (const float*)d_in[23];
  const float* w_gate = (const float*)d_in[24];
  const float* we1    = (const float*)d_in[25];
  const float* be1    = (const float*)d_in[26];
  const float* we2    = (const float*)d_in[27];
  const float* be2    = (const float*)d_in[28];
  const float* ffa_g  = (const float*)d_in[29];
  const float* ffa_w1 = (const float*)d_in[30];
  const float* ffa_b1 = (const float*)d_in[31];
  const float* ffa_w2 = (const float*)d_in[32];
  const float* ffa_b2 = (const float*)d_in[33];
  float* outp = (float*)d_out;

  // ---------------- workspace arena ----------------
  char* base = (char*)d_ws;
  size_t off = 0;
  auto allocF = [&](size_t n) -> float* {
    float* p = (float*)(base + off); off += ((n * 4 + 255) / 256) * 256; return p;
  };
  auto allocH = [&](size_t n) -> ushort_t* {
    ushort_t* p = (ushort_t*)(base + off); off += ((n * 2 + 255) / 256) * 256; return p;
  };
  // fp32 buffers
  float* x    = allocF((size_t)TOK * DM);
  float* hbuf = allocF((size_t)TOK * DM);
  float* sc   = allocF((size_t)BBS * HN * SQ * SQ);
  float* ybuf = allocF((size_t)TOK * DM);
  float* tbig = allocF((size_t)TOK * GH);
  float* x1   = allocF((size_t)TOK * DM);
  float* lg8  = allocF((size_t)TOK * EXPN);
  float* et   = allocF((size_t)BBS * CAPC * GH);
  float* eout = allocF((size_t)EXPN * BBS * CAPC * DM);
  float* auxb = allocF(32);                  // [0..7]=m1cnt [8..15]=psum [16]=z [17]=aux
  int* ie1 = (int*)allocF(TOK);
  int* ip1 = (int*)allocF(TOK);
  int* ie2 = (int*)allocF(TOK);
  int* ip2 = (int*)allocF(TOK);
  float* fg1 = allocF(TOK);
  float* fg2 = allocF(TOK);
  // bf16 activation buffers
  ushort_t* hb16   = allocH((size_t)TOK * DM);
  ushort_t* zb16   = allocH((size_t)TOK * DM);
  ushort_t* qkv16  = allocH((size_t)TOK * 3 * DM);
  ushort_t* p16    = allocH((size_t)BBS * HN * SQ * SQ);
  ushort_t* ao16   = allocH((size_t)TOK * DM);
  ushort_t* act16  = allocH((size_t)TOK * FFD);
  ushort_t* gg16   = allocH((size_t)TOK * HHID);
  ushort_t* xin16  = allocH((size_t)EXPN * BBS * CAPC * DM);
  ushort_t* eh16   = allocH((size_t)BBS * CAPC * HHID);
  // bf16 weights (converted once per launch)
  ushort_t* wqkv16 = allocH((size_t)6 * DM * 3 * DM);
  ushort_t* wao16  = allocH((size_t)6 * DM * DM);
  ushort_t* wfc16  = allocH((size_t)6 * DM * FFD);
  ushort_t* wmp16  = allocH((size_t)6 * FFD * DM);
  ushort_t* fb1_16 = allocH((size_t)DM * GH);
  ushort_t* fb2_16 = allocH((size_t)HHID * DM);
  ushort_t* fa1_16 = allocH((size_t)DM * GH);
  ushort_t* fa2_16 = allocH((size_t)HHID * DM);
  ushort_t* wg16   = allocH((size_t)DM * EXPN);
  ushort_t* we1_16 = allocH((size_t)EXPN * DM * GH);
  ushort_t* we2_16 = allocH((size_t)EXPN * HHID * DM);
  ushort_t* wte16  = allocH((size_t)VV * DM);

  const long ND = (long)TOK * DM;
  auto cvt = [&](const float* s, ushort_t* d, long n) {
    k_f2bf<<<eg(n), 256, 0, stream>>>(s, d, n);
  };

  // ---- one-time weight conversion to bf16 ----
  cvt(w_qkv, wqkv16, (long)6 * DM * 3 * DM);
  cvt(w_ao,  wao16,  (long)6 * DM * DM);
  cvt(w_fc,  wfc16,  (long)6 * DM * FFD);
  cvt(w_mp,  wmp16,  (long)6 * FFD * DM);
  cvt(ffb_w1, fb1_16, (long)DM * GH);
  cvt(ffb_w2, fb2_16, (long)HHID * DM);
  cvt(ffa_w1, fa1_16, (long)DM * GH);
  cvt(ffa_w2, fa2_16, (long)HHID * DM);
  cvt(w_gate, wg16, (long)DM * EXPN);
  cvt(we1, we1_16, (long)EXPN * DM * GH);
  cvt(we2, we2_16, (long)EXPN * HHID * DM);
  cvt(wte, wte16, (long)VV * DM);

  k_zero<<<1, 256, 0, stream>>>(auxb + 17, 1);
  embed_kernel<<<TOK, 256, 0, stream>>>(ids, wte, wpe, x);

  auto run_moe_block = [&]() {
    // x1 = h + geglu_ff(rms(h, ffb_g))
    norm_kernel<<<TOK, 256, 0, stream>>>(hbuf, ffb_g, nullptr, nullptr, zb16, DM, 1);
    gemm_launch(stream, false, zb16, fb1_16, ffb_b1, tbig, TOK, GH, DM, DM, GH, GH);
    k_geglu<<<eg((long)TOK * HHID), 256, 0, stream>>>(tbig, gg16, TOK);
    gemm_launch(stream, false, gg16, fb2_16, ffb_b2, ybuf, TOK, DM, HHID, HHID, DM, DM);
    k_add2<<<eg(ND), 256, 0, stream>>>(x1, hbuf, ybuf, ND);
    // gating on rms(x1, moe_g)
    norm_kernel<<<TOK, 256, 0, stream>>>(x1, moe_g, nullptr, nullptr, zb16, DM, 1);
    gemm_launch(stream, false, zb16, wg16, nullptr, lg8, TOK, EXPN, DM, DM, EXPN, EXPN);
    k_zero<<<1, 256, 0, stream>>>(auxb, 17);
    moe_gate<<<BBS, 32, 0, stream>>>(lg8, ie1, ip1, fg1, ie2, ip2, fg2,
                                     auxb, auxb + 8, auxb + 16);
    k_zero<<<eg((long)EXPN * BBS * CAPC * DM / 2), 256, 0, stream>>>(
        (float*)xin16, (long)EXPN * BBS * CAPC * DM / 2);
    moe_scatter<<<TOK, 256, 0, stream>>>(zb16, ie1, ip1, ie2, ip2, fg2, xin16);
    // experts (sequential to bound workspace)
    for (int e = 0; e < EXPN; ++e) {
      gemm_launch(stream, false,
                  xin16 + (size_t)e * BBS * CAPC * DM,
                  we1_16 + (size_t)e * DM * GH, be1 + (size_t)e * GH, et,
                  BBS * CAPC, GH, DM, DM, GH, GH);
      k_geglu<<<eg((long)BBS * CAPC * HHID), 256, 0, stream>>>(et, eh16, BBS * CAPC);
      gemm_launch(stream, false, eh16,
                  we2_16 + (size_t)e * HHID * DM, be2 + (size_t)e * DM,
                  eout + (size_t)e * BBS * CAPC * DM,
                  BBS * CAPC, DM, HHID, HHID, DM, DM);
    }
    moe_gather<<<TOK, 256, 0, stream>>>(x1, eout, ie1, ip1, fg1, ie2, ip2, fg2);
    // x1 += geglu_ff(rms(x1, ffa_g))
    norm_kernel<<<TOK, 256, 0, stream>>>(x1, ffa_g, nullptr, nullptr, zb16, DM, 1);
    gemm_launch(stream, false, zb16, fa1_16, ffa_b1, tbig, TOK, GH, DM, DM, GH, GH);
    k_geglu<<<eg((long)TOK * HHID), 256, 0, stream>>>(tbig, gg16, TOK);
    gemm_launch(stream, false, gg16, fa2_16, ffa_b2, ybuf, TOK, DM, HHID, HHID, DM, DM);
    k_add<<<eg(ND), 256, 0, stream>>>(x1, ybuf, ND);
    // outer residual + aux
    k_add<<<eg(ND), 256, 0, stream>>>(x, x1, ND);
    moe_aux<<<1, 32, 0, stream>>>(auxb, auxb + 8, auxb + 16, auxb + 17);
  };

  for (int i = 0; i < 6; ++i) {
    // ---- attention ----
    norm_kernel<<<TOK, 256, 0, stream>>>(x, ln1_g + (size_t)i * DM, ln1_b + (size_t)i * DM,
                                         nullptr, hb16, DM, 0);
    gemm_launch(stream, false, hb16, wqkv16 + (size_t)i * DM * 3 * DM,
                b_qkv + (size_t)i * 3 * DM, qkv16, TOK, 3 * DM, DM, DM, 3 * DM, 3 * DM,
                1, 1, 0, 0, 0, 0, 0, 0, 0, /*outbf=*/1);
    // scores[b,h] = q @ k^T  (A@B^T, K=64), batched over (b,h)
    gemm_launch(stream, true, qkv16, qkv16 + DM, nullptr, sc,
                SQ, SQ, DHD, 3 * DM, 3 * DM, SQ, BBS, HN,
                (long)SQ * 3 * DM, DHD, (long)SQ * 3 * DM, DHD,
                (long)HN * SQ * SQ, (long)SQ * SQ);
    attn_softmax<<<BBS * HN * SQ, 256, 0, stream>>>(sc, p16);
    // o[b,:,h*64..] = p @ v  (bf16 out)
    gemm_launch(stream, false, p16, qkv16 + 2 * DM, nullptr, ao16,
                SQ, DHD, SQ, SQ, 3 * DM, DM, BBS, HN,
                (long)HN * SQ * SQ, (long)SQ * SQ, (long)SQ * 3 * DM, DHD,
                (long)SQ * DM, DHD, 0, /*outbf=*/1);
    gemm_launch(stream, false, ao16, wao16 + (size_t)i * DM * DM, b_ao + (size_t)i * DM,
                ybuf, TOK, DM, DM, DM, DM, DM);
    k_add<<<eg(ND), 256, 0, stream>>>(x, ybuf, ND);
    // ---- FF / MoE ----
    norm_kernel<<<TOK, 256, 0, stream>>>(x, ln2_g + (size_t)i * DM, ln2_b + (size_t)i * DM,
                                         hbuf, hb16, DM, 0);
    if (i == 0 || i == 3) {
      run_moe_block();
    } else {
      // gelu fused into first GEMM epilogue, bf16 out
      gemm_launch(stream, false, hb16, wfc16 + (size_t)i * DM * FFD,
                  b_fc + (size_t)i * FFD, act16, TOK, FFD, DM, DM, FFD, FFD,
                  1, 1, 0, 0, 0, 0, 0, 0, /*actmode=*/1, /*outbf=*/1);
      gemm_launch(stream, false, act16, wmp16 + (size_t)i * FFD * DM,
                  b_mp + (size_t)i * DM, ybuf, TOK, DM, FFD, FFD, DM, DM);
      k_add<<<eg(ND), 256, 0, stream>>>(x, ybuf, ND);
    }
  }

  // final LN + tied LM head (A @ wte^T)
  norm_kernel<<<TOK, 256, 0, stream>>>(x, lnf_g, lnf_b, nullptr, hb16, DM, 0);
  gemm_launch(stream, true, hb16, wte16, nullptr, outp, TOK, VV, DM, DM, DM, VV);
  k_set<<<1, 32, 0, stream>>>(outp + (size_t)TOK * VV, auxb + 17);
}
